// MoD_2293512536086
// MI455X (gfx1250) — compile-verified, matrix operand order checked by reading, not run on hardware
//
#include <hip/hip_runtime.h>

// ---------------- problem constants ----------------
#define B_   4
#define S_   8192
#define D_   1024
#define H_   4096
#define KSEL 4096          // CAPACITY * S_

// ---------------- WMMA vector types ----------------
typedef __attribute__((ext_vector_type(16))) __bf16 v16bf;
typedef __attribute__((ext_vector_type(8)))  float  v8f;
typedef __attribute__((ext_vector_type(4)))  unsigned int u32x4;

union FragBF16 { u32x4 u[2]; v16bf v; };

// float -> bf16 round-to-nearest-even
__device__ __forceinline__ unsigned short f2bf(float f) {
    unsigned int u = __float_as_uint(f);
    u += 0x7FFFu + ((u >> 16) & 1u);
    return (unsigned short)(u >> 16);
}

// ---------------------------------------------------------------------------
// Kernel 1: tiled transpose + fp32->bf16.  src: float [R][C] -> dst: bf16 [C][R]
// ---------------------------------------------------------------------------
__global__ __launch_bounds__(256) void transpose_bf16_kernel(
    const float* __restrict__ src, unsigned short* __restrict__ dst, int R, int C)
{
    __shared__ unsigned short tile[32][33];
    int tilesC = C >> 5;
    int tc = blockIdx.x % tilesC;
    int tr = blockIdx.x / tilesC;
    int c0 = tc << 5, r0 = tr << 5;
    int tx = threadIdx.x & 31, ty = threadIdx.x >> 5;  // ty in 0..7
#pragma unroll
    for (int i = 0; i < 4; ++i) {
        int r = r0 + ty + i * 8;
        tile[ty + i * 8][tx] = f2bf(src[(size_t)r * C + c0 + tx]);
    }
    __syncthreads();
#pragma unroll
    for (int i = 0; i < 4; ++i) {
        int c = c0 + ty + i * 8;
        dst[(size_t)c * R + r0 + tx] = tile[tx][ty + i * 8];
    }
}

// ---------------------------------------------------------------------------
// Kernel 2: router scores + x -> bf16 conversion.  One wave32 per token.
// ---------------------------------------------------------------------------
__global__ __launch_bounds__(256) void router_kernel(
    const float* __restrict__ x, const float* __restrict__ wr,
    const float* __restrict__ br,
    unsigned short* __restrict__ xbf, float* __restrict__ weights)
{
    int wv = threadIdx.x >> 5, lane = threadIdx.x & 31;
    int tok = blockIdx.x * 8 + wv;                     // 0 .. B*S-1
    const float* xr = x + (size_t)tok * D_;
    unsigned short* xo = xbf + (size_t)tok * D_;
    float acc = 0.f;
#pragma unroll
    for (int i = 0; i < 8; ++i) {
        int off = lane * 4 + i * 128;
        float4 v = *(const float4*)(xr + off);
        float4 w = *(const float4*)(wr + off);
        acc += v.x * w.x + v.y * w.y + v.z * w.z + v.w * w.w;
        union { unsigned short s[4]; uint2 u; } p;
        p.s[0] = f2bf(v.x); p.s[1] = f2bf(v.y);
        p.s[2] = f2bf(v.z); p.s[3] = f2bf(v.w);
        *(uint2*)(xo + off) = p.u;
    }
#pragma unroll
    for (int m = 16; m; m >>= 1) acc += __shfl_xor(acc, m, 32);
    if (lane == 0) weights[tok] = acc + br[0];
}

// ---------------------------------------------------------------------------
// Kernel 3: exact k-th largest per batch via 32-pass bitwise radix select.
// Also zero-fills counts and the index list (safe padding for the MLP gather).
// ---------------------------------------------------------------------------
__global__ __launch_bounds__(1024) void select_kernel(
    const float* __restrict__ weights, float* __restrict__ thr,
    int* __restrict__ counts, int* __restrict__ indices)
{
    __shared__ unsigned int keys[S_];
    __shared__ unsigned int cnt_s;
    int bb = blockIdx.x, tid = threadIdx.x;
    for (int i = tid; i < S_; i += 1024) {
        unsigned int u = __float_as_uint(weights[bb * S_ + i]);
        keys[i] = (u & 0x80000000u) ? ~u : (u | 0x80000000u);  // order-preserving
        indices[bb * S_ + i] = 0;
    }
    __syncthreads();
    unsigned int prefix = 0; int kk = KSEL;
    for (int bit = 31; bit >= 0; --bit) {
        if (tid == 0) cnt_s = 0;
        __syncthreads();
        unsigned int tgt = (prefix >> bit) | 1u;   // candidate high bits
        int local = 0;
        for (int i = tid; i < S_; i += 1024)
            if ((keys[i] >> bit) == tgt) local++;
        if (local) atomicAdd(&cnt_s, (unsigned int)local);
        __syncthreads();
        unsigned int c = cnt_s;
        __syncthreads();
        if ((int)c >= kk) prefix |= (1u << bit);
        else              kk -= (int)c;
    }
    if (tid == 0) {
        unsigned int u = (prefix & 0x80000000u) ? (prefix & 0x7FFFFFFFu) : ~prefix;
        thr[bb] = __uint_as_float(u);
        counts[bb] = 0;
    }
}

// ---------------------------------------------------------------------------
// Kernel 4a: compact indices of selected tokens (weights > threshold, strict).
// ---------------------------------------------------------------------------
__global__ __launch_bounds__(256) void compact_kernel(
    const float* __restrict__ weights, const float* __restrict__ thr,
    int* __restrict__ counts, int* __restrict__ indices)
{
    int t = blockIdx.x * 256 + threadIdx.x;
    if (t >= B_ * S_) return;
    int bb = t / S_;
    if (weights[t] > thr[bb]) {
        int pos = atomicAdd(&counts[bb], 1);
        indices[bb * S_ + pos] = t % S_;
    }
}

// ---------------------------------------------------------------------------
// Kernel 4b: identity passthrough for unselected tokens (wave32 per token).
// ---------------------------------------------------------------------------
__global__ __launch_bounds__(256) void passthrough_kernel(
    const float* __restrict__ x, const float* __restrict__ weights,
    const float* __restrict__ thr, float* __restrict__ out)
{
    int wv = threadIdx.x >> 5, lane = threadIdx.x & 31;
    int t = blockIdx.x * 8 + wv;
    int bb = t / S_;
    if (weights[t] > thr[bb]) return;      // MLP kernel writes these rows
    const float4* src = (const float4*)(x + (size_t)t * D_);
    float4* dst = (float4*)(out + (size_t)t * D_);
#pragma unroll
    for (int i = 0; i < 8; ++i) dst[lane + 32 * i] = src[lane + 32 * i];
}

// ---------------------------------------------------------------------------
// Kernel 5: fused MLP over selected tokens.
//   64-token tile / workgroup, 16 waves. x-tile resident in LDS (132 KB),
//   staged via CDNA5 async global->LDS copies (ASYNCcnt-tracked).
//   Per 64-wide h-chunk: StageA H = gelu(x@W1+b1) via WMMA -> bf16 LDS tile,
//   StageB out += H @ W2 accumulated in VGPRs. Hidden never hits global mem.
// ---------------------------------------------------------------------------
#define BM      64
#define XSTR    (D_ + 8)      // 1032 bf16, padded to dodge bank conflicts
#define HC      64
#define HSTR    (HC + 8)      // 72 bf16

__global__ __launch_bounds__(512) void mlp_kernel(
    const unsigned short* __restrict__ xbf,   // [B*S][D] bf16
    const unsigned short* __restrict__ w1t,   // [H][D]  bf16  (W1 transposed)
    const unsigned short* __restrict__ w2t,   // [D][H]  bf16  (W2 transposed)
    const float* __restrict__ b1, const float* __restrict__ b2,
    const int* __restrict__ indices, const int* __restrict__ counts,
    float* __restrict__ out)
{
    __shared__ unsigned short xt[BM * XSTR];
    __shared__ unsigned short ht[BM * HSTR];
    __shared__ int toks[BM];

    const int tilesPerBatch = KSEL / BM;               // 64
    int bb   = blockIdx.x / tilesPerBatch;
    int row0 = (blockIdx.x % tilesPerBatch) * BM;
    int cnt  = counts[bb];
    int tid  = threadIdx.x;

    if (tid < BM) {
        int r = row0 + tid;
        int rr = (r < cnt) ? r : (cnt > 0 ? cnt - 1 : 0);   // clamp (safe gather)
        toks[tid] = indices[bb * S_ + rr];
    }
    __syncthreads();

    // stage x tile into LDS: 64 rows x 1024 bf16, 16B chunks.
    // CDNA5 direct async global->LDS (no VGPR round-trip); LDS operand is the
    // low 32 bits of the flat shared address (ISA 10.2 aperture truncation).
    for (int c = tid; c < BM * (D_ / 8); c += 512) {
        int r = c / (D_ / 8), kc = c % (D_ / 8);
        const unsigned short* gp =
            xbf + ((size_t)bb * S_ + toks[r]) * D_ + kc * 8;
        unsigned int lp = (unsigned int)(size_t)&xt[r * XSTR + kc * 8];
        asm volatile("global_load_async_to_lds_b128 %0, %1, off"
                     :: "v"(lp), "v"(gp) : "memory");
    }
    asm volatile("s_wait_asynccnt 0" ::: "memory");
    __syncthreads();

    const int wv   = tid >> 5;          // 0..15
    const int lane = tid & 31;
    const int l16  = lane & 15;
    const int hi   = lane >> 4;         // lane half selects K sub-pattern
    const int wm   = wv & 3;            // out-row group  (16*wm)
    const int wn   = wv >> 2;           // out-col group  (256*wn)

    // output accumulators: 16 tiles of 16x16 = rows [16wm,16wm+16) x cols [256wn,256wn+256)
    v8f oacc[16];
#pragma unroll
    for (int t = 0; t < 16; ++t) {
        float bv = b2[256 * wn + 16 * t + l16];
#pragma unroll
        for (int j = 0; j < 8; ++j) oacc[t][j] = bv;
    }

    for (int hc = 0; hc < H_; hc += HC) {
        // ---- Stage A: H[64x64] = gelu(x_tile @ W1[:,hc:hc+64] + b1) ----
        v8f hacc;
        {
            float bv = b1[hc + 16 * wn + l16];
#pragma unroll
            for (int j = 0; j < 8; ++j) hacc[j] = bv;
        }
        const int arow = 16 * wm + l16;
        for (int k = 0; k < D_; k += 32) {
            int kb = k + hi * 8;
            FragBF16 fa, fb;
            fa.u[0] = *(const u32x4*)&xt[arow * XSTR + kb];
            fa.u[1] = *(const u32x4*)&xt[arow * XSTR + kb + 16];
            const unsigned short* bp =
                w1t + (size_t)(hc + 16 * wn + l16) * D_ + kb;
            __builtin_prefetch(bp + 64, 0, 0);
            fb.u[0] = *(const u32x4*)bp;
            fb.u[1] = *(const u32x4*)(bp + 16);
            hacc = __builtin_amdgcn_wmma_f32_16x16x32_bf16(
                false, fa.v, false, fb.v, (short)0, hacc, false, false);
        }
        // exact GELU, write bf16 H tile to LDS
#pragma unroll
        for (int j = 0; j < 8; ++j) {
            float v = hacc[j];
            float g = 0.5f * v * (1.0f + erff(v * 0.70710678118654752f));
            int r = 16 * wm + (hi ? j + 8 : j);
            ht[r * HSTR + 16 * wn + l16] = f2bf(g);
        }
        __syncthreads();

        // ---- Stage B: oacc += H[64x64] @ W2[hc:hc+64, :] ----
#pragma unroll
        for (int kk = 0; kk < HC; kk += 32) {
            int kb = kk + hi * 8;
            FragBF16 fa;
            fa.u[0] = *(const u32x4*)&ht[(16 * wm + l16) * HSTR + kb];
            fa.u[1] = *(const u32x4*)&ht[(16 * wm + l16) * HSTR + kb + 16];
#pragma unroll
            for (int t = 0; t < 16; ++t) {
                FragBF16 fb;
                const unsigned short* bp =
                    w2t + (size_t)(256 * wn + 16 * t + l16) * H_ + hc + kb;
                fb.u[0] = *(const u32x4*)bp;
                fb.u[1] = *(const u32x4*)(bp + 16);
                oacc[t] = __builtin_amdgcn_wmma_f32_16x16x32_bf16(
                    false, fa.v, false, fb.v, (short)0, oacc[t], false, false);
            }
        }
        __syncthreads();   // H tile consumed; safe to overwrite next chunk
    }

    // ---- epilogue: scatter fp32 rows of selected tokens ----
#pragma unroll
    for (int t = 0; t < 16; ++t) {
        int ccol = 256 * wn + 16 * t + l16;
#pragma unroll
        for (int j = 0; j < 8; ++j) {
            int r = 16 * wm + (hi ? j + 8 : j);
            if (row0 + r < cnt) {
                int s_idx = toks[r];
                out[((size_t)bb * S_ + s_idx) * D_ + ccol] = oacc[t][j];
            }
        }
    }
}

// ---------------------------------------------------------------------------
// launch
// ---------------------------------------------------------------------------
extern "C" void kernel_launch(void* const* d_in, const int* in_sizes, int n_in,
                              void* d_out, int out_size, void* d_ws, size_t ws_size,
                              hipStream_t stream) {
    (void)in_sizes; (void)n_in; (void)out_size; (void)ws_size;
    const float* x  = (const float*)d_in[0];
    const float* wr = (const float*)d_in[1];
    const float* br = (const float*)d_in[2];
    const float* W1 = (const float*)d_in[3];
    const float* b1 = (const float*)d_in[4];
    const float* W2 = (const float*)d_in[5];
    const float* b2 = (const float*)d_in[6];
    float* out = (float*)d_out;

    char* ws = (char*)d_ws;
    unsigned short* xbf     = (unsigned short*)(ws);                 // 64 MB
    unsigned short* w1t     = (unsigned short*)(ws + 67108864);      // 8 MB
    unsigned short* w2t     = (unsigned short*)(ws + 75497472);      // 8 MB
    float*          weights = (float*)(ws + 83886080);               // 128 KB
    float*          thr     = (float*)(ws + 84017152);
    int*            counts  = (int*)(ws + 84017408);
    int*            indices = (int*)(ws + 84017664);                 // 128 KB

    transpose_bf16_kernel<<<(D_ / 32) * (H_ / 32), 256, 0, stream>>>(W1, w1t, D_, H_);
    transpose_bf16_kernel<<<(H_ / 32) * (D_ / 32), 256, 0, stream>>>(W2, w2t, H_, D_);
    router_kernel<<<(B_ * S_) / 8, 256, 0, stream>>>(x, wr, br, xbf, weights);
    select_kernel<<<B_, 1024, 0, stream>>>(weights, thr, counts, indices);
    compact_kernel<<<(B_ * S_) / 256, 256, 0, stream>>>(weights, thr, counts, indices);
    passthrough_kernel<<<(B_ * S_) / 8, 256, 0, stream>>>(x, weights, thr, out);
    mlp_kernel<<<B_ * (KSEL / BM), 512, 0, stream>>>(xbf, w1t, w2t, b1, b2,
                                                     indices, counts, out);
}
